// AttnBlock_24464133718510
// MI455X (gfx1250) — matrix operand order repaired and checked by reading, not compiled, not run on hardware
//
#include <hip/hip_runtime.h>

typedef __attribute__((ext_vector_type(16))) _Float16 v16h;
typedef __attribute__((ext_vector_type(8)))  _Float16 v8h;
typedef __attribute__((ext_vector_type(8)))  float    v8f;
typedef __attribute__((ext_vector_type(4)))  int      v4i;

#define BB 4
#define CC 256
#define LL 4096
#define CPG 8   // channels per group (C / 32 groups)

// ---- gfx1250 async global->LDS path (guarded; sync copy fallback) ----
#if defined(__has_builtin)
#  if __has_builtin(__builtin_amdgcn_global_load_async_to_lds_b128)
#    define USE_ASYNC_LDS 1
#  endif
#endif
#ifndef USE_ASYNC_LDS
#  define USE_ASYNC_LDS 0
#endif

#if USE_ASYNC_LDS
typedef __attribute__((address_space(1))) v4i gv4i_t;   // global int4
typedef __attribute__((address_space(3))) v4i lv4i_t;   // LDS int4
__device__ __forceinline__ void async_ld_b128(const _Float16* gsrc, _Float16* ldst) {
  __builtin_amdgcn_global_load_async_to_lds_b128((gv4i_t*)gsrc, (lv4i_t*)ldst, 0, 0);
}
#endif

__device__ __forceinline__ void wait_async_lds() {
#if USE_ASYNC_LDS
#  if __has_builtin(__builtin_amdgcn_s_wait_asynccnt)
  __builtin_amdgcn_s_wait_asynccnt(0);
#  else
  asm volatile("s_wait_asynccnt 0x0" ::: "memory");
#  endif
#endif
}

// ---------------- WMMA helpers (gfx1250 wave32 layouts) ----------------
// A operand 16x32 f16: lane = (h<<4)|r ; row M=r ; K chunks [8h,8h+8) and [16+8h,16+8h+8)
__device__ __forceinline__ v16h load_a16(const _Float16* p, int ld) {
  const int lane = threadIdx.x & 31;
  const int r = lane & 15, h = lane >> 4;
  const _Float16* q = p + (size_t)r * ld + 8 * h;
  v8h lo = *(const v8h*)(q);
  v8h hi = *(const v8h*)(q + 16);
  v16h out;
#pragma unroll
  for (int i = 0; i < 8; ++i) { out[i] = lo[i]; out[8 + i] = hi[i]; }
  return out;
}

// B operand 32x16 f16: lane = (h<<4)|n ; col N=n ; K chunks [8h,8h+8) and [16+8h,16+8h+8)
// Source must be laid out [n][k] with k contiguous (row stride ld elements).
__device__ __forceinline__ v16h load_b16(const _Float16* p, int ld) {
  const int lane = threadIdx.x & 31;
  const int n = lane & 15, h = lane >> 4;
  const _Float16* q = p + (size_t)n * ld + 8 * h;
  v8h lo = *(const v8h*)(q);
  v8h hi = *(const v8h*)(q + 16);
  v16h out;
#pragma unroll
  for (int i = 0; i < 8; ++i) { out[i] = lo[i]; out[8 + i] = hi[i]; }
  return out;
}

__device__ __forceinline__ v8f wmma32(v16h a, v16h b, v8f c) {
  return __builtin_amdgcn_wmma_f32_16x16x32_f16(false, a, false, b, (short)0, c,
                                                false, false);
}

// ---------------- 0. convert all four weight matrices to f16 ----------------
__global__ __launch_bounds__(256) void cvt_w_kernel(const float* qw, const float* kw,
                                                    const float* vw, const float* pw,
                                                    _Float16* out) {
  int i = blockIdx.x * 256 + threadIdx.x;      // 4 * 65536 threads
  int m = i >> 16, j = i & 0xFFFF;
  const float* src = (m == 0) ? qw : (m == 1) ? kw : (m == 2) ? vw : pw;
  out[i] = (_Float16)src[j];
}

// ---------------- 1. GroupNorm statistics ----------------
__global__ __launch_bounds__(256) void gn_stats_kernel(const float* x, float* mean,
                                                       float* rstd) {
  int bg = blockIdx.x;                 // b*32 + g
  int b = bg >> 5, g = bg & 31;
  const float* base = x + ((size_t)b * CC + g * CPG) * LL;
  float s = 0.f, ss = 0.f;
  for (int c = 0; c < CPG; ++c)
    for (int l = threadIdx.x; l < LL; l += 256) {
      float v = base[(size_t)c * LL + l];
      s += v; ss += v * v;
    }
  __shared__ float s1[256], s2[256];
  s1[threadIdx.x] = s; s2[threadIdx.x] = ss;
  __syncthreads();
  for (int off = 128; off > 0; off >>= 1) {
    if (threadIdx.x < off) {
      s1[threadIdx.x] += s1[threadIdx.x + off];
      s2[threadIdx.x] += s2[threadIdx.x + off];
    }
    __syncthreads();
  }
  if (threadIdx.x == 0) {
    const float invN = 1.0f / (float)(CPG * LL);
    float mu = s1[0] * invN;
    float var = s2[0] * invN - mu * mu;
    mean[bg] = mu;
    rstd[bg] = rsqrtf(var + 1e-6f);
  }
}

// ---------------- 2. apply GroupNorm, write h_t[b][l][c] (f16, transposed) --------
__global__ __launch_bounds__(256) void gn_apply_kernel(const float* x, const float* gam,
                                                       const float* bet, const float* mean,
                                                       const float* rstd, _Float16* ht) {
  __shared__ _Float16 sT[32][72];      // 32 c x 64 l tile (+pad)
  int b = blockIdx.z, c0 = blockIdx.y * 32, l0 = blockIdx.x * 64;
  int tid = threadIdx.x;
  int ll = tid & 63, cc = tid >> 6;    // cc in [0,4)
#pragma unroll
  for (int r = 0; r < 8; ++r) {
    int cl = r * 4 + cc;
    int c = c0 + cl;
    int bg = b * 32 + (c >> 3);
    float mu = mean[bg], rs = rstd[bg];
    float v = x[((size_t)b * CC + c) * LL + l0 + ll];
    sT[cl][ll] = (_Float16)((v - mu) * rs * gam[c] + bet[c]);
  }
  __syncthreads();
  int c2 = tid & 31, lw = tid >> 5;    // lw in [0,8)
#pragma unroll
  for (int r = 0; r < 8; ++r) {
    int lloc = r * 8 + lw;
    ht[((size_t)b * LL + l0 + lloc) * CC + c0 + c2] = sT[c2][lloc];
  }
}

// ------- 3. Q/K/V GEMMs: wave holds one 16-l B tile, sweeps 4 o-tiles x 3 mats ------
__global__ __launch_bounds__(128) void qkv_kernel(const _Float16* ht, const _Float16* wbuf,
                                                  const float* qb, const float* kb,
                                                  const float* vb,
                                                  _Float16* qt, _Float16* kt, _Float16* vv) {
  const int w = blockIdx.x * 4 + (threadIdx.x >> 5);   // 0..4095
  const int lane = threadIdx.x & 31;
  const int n = lane & 15, h = lane >> 4;
  const int og = (w & 3) * 64;               // o-tile group base
  const int l0 = ((w >> 2) & 255) * 16;
  const int b = w >> 10;

  const _Float16* wq = wbuf;
  const _Float16* wk = wbuf + 65536;
  const _Float16* wv = wbuf + 131072;
  const _Float16* hb = ht + ((size_t)b * LL + l0) * CC;

  v16h bop[8];                               // whole 16l x 256c B tile resident
#pragma unroll
  for (int kc = 0; kc < 8; ++kc) bop[kc] = load_b16(hb + kc * 32, CC);

  const float qsc = 0.0625f;                 // C^-0.5 folded into Q
  const int l = l0 + n;
#pragma unroll
  for (int t = 0; t < 4; ++t) {
    const int o0 = og + t * 16;
    v8f aq = {}, ak = {}, av = {};
#pragma unroll
    for (int kc = 0; kc < 8; ++kc) {
      v16h A;
      A = load_a16(wq + (size_t)o0 * CC + kc * 32, CC); aq = wmma32(A, bop[kc], aq);
      A = load_a16(wk + (size_t)o0 * CC + kc * 32, CC); ak = wmma32(A, bop[kc], ak);
      A = load_a16(wv + (size_t)o0 * CC + kc * 32, CC); av = wmma32(A, bop[kc], av);
    }
    v8f bq = *(const v8f*)(qb + o0 + 8 * h);
    v8f bk = *(const v8f*)(kb + o0 + 8 * h);
    v8f bv = *(const v8f*)(vb + o0 + 8 * h);
    v8h hq, hk;
#pragma unroll
    for (int p = 0; p < 8; ++p) {
      hq[p] = (_Float16)((aq[p] + bq[p]) * qsc);
      hk[p] = (_Float16)(ak[p] + bk[p]);
    }
    *(v8h*)(qt + ((size_t)b * LL + l) * CC + o0 + 8 * h) = hq;  // q_t[b][l][c]
    *(v8h*)(kt + ((size_t)b * LL + l) * CC + o0 + 8 * h) = hk;  // k_t[b][l][c]
#pragma unroll
    for (int p = 0; p < 8; ++p)                                  // v[b][c][l]
      vv[((size_t)b * CC + o0 + 8 * h + p) * LL + l] = (_Float16)(av[p] + bv[p]);
  }
}

// ---------------- 4. flash attention ----------------
// WG = (b, 64 i-rows); wave = 16 i-rows.  K/V chunks double-buffered in LDS,
// staged with global_load_async_to_lds_b128 overlapping the WMMA compute.
__device__ __forceinline__ void stage_chunk(const _Float16* kbase, const _Float16* vbase,
                                            int j0, int tid,
                                            _Float16 (*sk)[264], _Float16 (*sv)[72]) {
#pragma unroll
  for (int it = 0; it < 16; ++it) {
    int idx = it * 128 + tid;
    int kr = idx >> 5, kc8 = (idx & 31) * 8;
    const _Float16* g = kbase + (size_t)(j0 + kr) * CC + kc8;
#if USE_ASYNC_LDS
    async_ld_b128(g, &sk[kr][kc8]);
#else
    *(v8h*)(&sk[kr][kc8]) = *(const v8h*)g;
#endif
  }
#pragma unroll
  for (int it = 0; it < 16; ++it) {
    int idx = it * 128 + tid;
    int vr = idx >> 3, vc8 = (idx & 7) * 8;
    const _Float16* g = vbase + (size_t)vr * LL + j0 + vc8;
#if USE_ASYNC_LDS
    async_ld_b128(g, &sv[vr][vc8]);
#else
    *(v8h*)(&sv[vr][vc8]) = *(const v8h*)g;
#endif
  }
}

__global__ __launch_bounds__(128) void attn_kernel(const _Float16* qt, const _Float16* kt,
                                                   const _Float16* vv, _Float16* ot) {
  __shared__ _Float16 sK[2][64][264];     // j-rows, c contiguous (+pad), double buffered
  __shared__ _Float16 sV[2][256][72];     // c-rows, j contiguous (+pad), double buffered
  __shared__ _Float16 sP[4][16][72];      // per-wave P transpose staging (+pad)

  const int tid = threadIdx.x;
  const int wave = tid >> 5, lane = tid & 31;
  const int n = lane & 15, h = lane >> 4;
  const int b = blockIdx.x >> 6;
  const int iw = (blockIdx.x & 63) * 64 + wave * 16;

  const _Float16* qbase = qt + ((size_t)b * LL + iw) * CC;
  const _Float16* kbase = kt + (size_t)b * LL * CC;
  const _Float16* vbase = vv + (size_t)b * CC * LL;

  const v8f vzero = {};
  v8f acc[16];
#pragma unroll
  for (int ct = 0; ct < 16; ++ct) acc[ct] = vzero;
  float mrow[8], lrow[8];
#pragma unroll
  for (int p = 0; p < 8; ++p) { mrow[p] = -3.0e38f; lrow[p] = 0.f; }

  stage_chunk(kbase, vbase, 0, tid, sK[0], sV[0]);
  wait_async_lds();
  __syncthreads();

  int cur = 0;
  for (int j0 = 0; j0 < LL; j0 += 64) {
    const int nxt = cur ^ 1;
    if (j0 + 64 < LL)                      // async-prefetch next chunk into other buffer
      stage_chunk(kbase, vbase, j0 + 64, tid, sK[nxt], sV[nxt]);

    // ---- S = (Q*scale)^T K : 4 tiles of 16j, K-dim = 256 ----
    v8f S[4];
#pragma unroll
    for (int jt = 0; jt < 4; ++jt) S[jt] = vzero;
#pragma unroll
    for (int kc = 0; kc < 8; ++kc) {
      v16h a = load_a16(qbase + kc * 32, CC);
#pragma unroll
      for (int jt = 0; jt < 4; ++jt) {
        v16h bop = load_b16(&sK[cur][jt * 16][kc * 32], 264);
        S[jt] = wmma32(a, bop, S[jt]);
      }
    }

    // ---- online softmax per row (row m = 8h+p lives in VGPR p across a lane half)
#pragma unroll
    for (int p = 0; p < 8; ++p) {
      float mx = fmaxf(fmaxf(S[0][p], S[1][p]), fmaxf(S[2][p], S[3][p]));
#pragma unroll
      for (int d = 1; d < 16; d <<= 1) mx = fmaxf(mx, __shfl_xor(mx, d, 32));
      float Mn = fmaxf(mrow[p], mx);
      float corr = __expf(mrow[p] - Mn);
      mrow[p] = Mn;
      float rs = 0.f;
#pragma unroll
      for (int jt = 0; jt < 4; ++jt) {
        float e = __expf(S[jt][p] - Mn);
        S[jt][p] = e;
        rs += e;
      }
#pragma unroll
      for (int d = 1; d < 16; d <<= 1) rs += __shfl_xor(rs, d, 32);
      lrow[p] = lrow[p] * corr + rs;
#pragma unroll
      for (int ct = 0; ct < 16; ++ct) acc[ct][p] *= corr;
    }

    // ---- transpose P (D layout -> A layout) through per-wave LDS tile ----
#pragma unroll
    for (int jt = 0; jt < 4; ++jt)
#pragma unroll
      for (int p = 0; p < 8; ++p)
        sP[wave][8 * h + p][jt * 16 + n] = (_Float16)S[jt][p];
    // same-wave DS ops are in-order; compiler inserts s_wait_dscnt before reuse

    // ---- O += P * V^T : 16 c-tiles x 2 j32-chunks ----
#pragma unroll
    for (int jh = 0; jh < 2; ++jh) {
      v16h a = load_a16(&sP[wave][0][jh * 32], 72);
#pragma unroll
      for (int ct = 0; ct < 16; ++ct) {
        v16h bop = load_b16(&sV[cur][ct * 16][jh * 32], 72);
        acc[ct] = wmma32(a, bop, acc[ct]);
      }
    }

    wait_async_lds();     // this wave's async loads into nxt complete
    __syncthreads();      // all waves done loading nxt + done reading cur
    cur = nxt;
  }

  // ---- normalize and store o_t[b][i][c] ----
  _Float16* obase = ot + ((size_t)b * LL + iw) * CC;
#pragma unroll
  for (int p = 0; p < 8; ++p) {
    float inv = 1.0f / lrow[p];
#pragma unroll
    for (int ct = 0; ct < 16; ++ct)
      obase[(size_t)(8 * h + p) * CC + ct * 16 + n] = (_Float16)(acc[ct][p] * inv);
  }
}

// ------- 5. projection + bias + residual: wave holds B tile, sweeps 4 o-tiles -------
__global__ __launch_bounds__(128) void proj_kernel(const _Float16* ot, const _Float16* wp,
                                                   const float* pb, const float* x,
                                                   float* out) {
  const int w = blockIdx.x * 4 + (threadIdx.x >> 5);   // 0..4095
  const int lane = threadIdx.x & 31;
  const int n = lane & 15, h = lane >> 4;
  const int og = (w & 3) * 64;
  const int l0 = ((w >> 2) & 255) * 16;
  const int b = w >> 10;

  const _Float16* ob = ot + ((size_t)b * LL + l0) * CC;
  v16h bop[8];
#pragma unroll
  for (int kc = 0; kc < 8; ++kc) bop[kc] = load_b16(ob + kc * 32, CC);

  const int l = l0 + n;
#pragma unroll
  for (int t = 0; t < 4; ++t) {
    const int o0 = og + t * 16;
    v8f a = {};
#pragma unroll
    for (int kc = 0; kc < 8; ++kc) {
      v16h A = load_a16(wp + (size_t)o0 * CC + kc * 32, CC);
      a = wmma32(A, bop[kc], a);
    }
    v8f bp = *(const v8f*)(pb + o0 + 8 * h);
#pragma unroll
    for (int p = 0; p < 8; ++p) {
      int o = o0 + 8 * h + p;
      size_t xi = ((size_t)b * CC + o) * LL + l;
      out[xi] = x[xi] + a[p] + bp[p];
    }
  }
}

// ---------------- host launcher ----------------
extern "C" void kernel_launch(void* const* d_in, const int* in_sizes, int n_in,
                              void* d_out, int out_size, void* d_ws, size_t ws_size,
                              hipStream_t stream) {
  const float* x  = (const float*)d_in[0];
  const float* ng = (const float*)d_in[1];
  const float* nb = (const float*)d_in[2];
  const float* qw = (const float*)d_in[3];
  const float* qb = (const float*)d_in[4];
  const float* kw = (const float*)d_in[5];
  const float* kb = (const float*)d_in[6];
  const float* vw = (const float*)d_in[7];
  const float* vb = (const float*)d_in[8];
  const float* pw = (const float*)d_in[9];
  const float* pb = (const float*)d_in[10];
  float* out = (float*)d_out;

  char* ws = (char*)d_ws;
  float* mean = (float*)ws;                    // 128 f32
  float* rstd = mean + 128;                    // 128 f32
  const size_t M = (size_t)BB * LL * CC;       // 4M elements
  _Float16* ht   = (_Float16*)(ws + 4096);     // h_t  [b][l][c]
  _Float16* qt   = ht + M;                     // q_t  [b][l][c] (scaled)
  _Float16* kt   = qt + M;                     // k_t  [b][l][c]
  _Float16* vv   = kt + M;                     // v    [b][c][l]
  _Float16* ob   = vv + M;                     // o_t  [b][l][c]
  _Float16* wbuf = ob + M;                     // f16 weights: q,k,v,proj

  cvt_w_kernel<<<1024, 256, 0, stream>>>(qw, kw, vw, pw, wbuf);
  gn_stats_kernel<<<128, 256, 0, stream>>>(x, mean, rstd);
  gn_apply_kernel<<<dim3(64, 8, 4), 256, 0, stream>>>(x, ng, nb, mean, rstd, ht);
  qkv_kernel<<<1024, 128, 0, stream>>>(ht, wbuf, qb, kb, vb, qt, kt, vv);
  attn_kernel<<<256, 128, 0, stream>>>(qt, kt, vv, ob);
  proj_kernel<<<1024, 128, 0, stream>>>(ob, wbuf + 3 * 65536, pb, x, out);
}
